// M_40879498724279
// MI455X (gfx1250) — matrix operand order repaired and checked by reading, not compiled, not run on hardware
//
#include <hip/hip_runtime.h>

// ---------------------------------------------------------------------------
// BN(inference) + ReLU + 1x1 conv  ==  out[o, n] = sum_c W[o,c] * relu(x*sc+sh)
// GEMM via V_WMMA_F32_16X16X32_BF16 with split-bf16 (hi+lo) for ~fp32 accuracy.
// W is pre-packed (split-bf16, A-fragment order) into workspace once, then the
// per-chunk W tile is DMA'd to LDS with the Tensor Data Mover when available.
// ---------------------------------------------------------------------------

typedef __attribute__((ext_vector_type(16))) __bf16       v16bf;
typedef __attribute__((ext_vector_type(8)))  float        v8f;
typedef __attribute__((ext_vector_type(4))) unsigned int  v4u;

#if defined(__has_builtin)
#if __has_builtin(__builtin_amdgcn_tensor_load_to_lds) && \
    __has_builtin(__builtin_amdgcn_s_wait_tensorcnt)
#define HAVE_TDM 1
typedef __attribute__((ext_vector_type(4))) unsigned int u32x4;
typedef __attribute__((ext_vector_type(8))) int          i32x8;
typedef __attribute__((ext_vector_type(4))) int          i32x4;
#endif
#endif

constexpr int BATCH = 256;
constexpr int CIN   = 2112;
constexpr int COUT  = 192;
constexpr int HW    = 49;                  // 7*7
constexpr int KC    = 32;                  // K per chunk (bf16 WMMA K)
constexpr int NCH   = CIN / KC;            // 66 chunks
constexpr int TN    = 128;                 // N columns per workgroup
constexpr int NWG   = (BATCH * HW) / TN;   // 98 (exact: 12544 = 98*128)
constexpr int MT    = COUT / 16;           // 12 M tiles
constexpr float EPS = 1e-5f;

constexpr int    WFRAG_WORDS  = MT * 32 * 8;          // 3072 words (one hi or lo plane)
constexpr int    WCHUNK_WORDS = WFRAG_WORDS * 2;      // 6144 words = 24576 B per chunk
constexpr size_t WS_NEEDED    = (size_t)NCH * WCHUNK_WORDS * 4;  // 1,622,016 B

union Frag {
    v16bf        v;
    v4u          q[2];
    unsigned int w[8];
};

__device__ inline unsigned int pack_bf16x2(__bf16 a, __bf16 b) {
    unsigned short au, bu;
    __builtin_memcpy(&au, &a, 2);
    __builtin_memcpy(&bu, &b, 2);
    return (unsigned int)au | ((unsigned int)bu << 16);
}

// Split two f32 values into packed (hi, lo) bf16 pairs: f ~= hi + lo.
__device__ inline void split2(float a, float b, unsigned int& hi, unsigned int& lo) {
    __bf16 ah = (__bf16)a, bh = (__bf16)b;                 // RNE
    __bf16 al = (__bf16)(a - (float)ah);
    __bf16 bl = (__bf16)(b - (float)bh);
    hi = pack_bf16x2(ah, bh);
    lo = pack_bf16x2(al, bl);
}

// 16-bit A 16x32 layout helper: VGPR v<4 -> K = half*8 + 2v ; v>=4 -> K = 16 + half*8 + 2(v-4)
__device__ inline int a_frag_kloc(int half, int v) {
    return (v < 4) ? (half * 8 + 2 * v) : (16 + half * 8 + 2 * (v - 4));
}

// ---------------------------------------------------------------------------
// Prep: pack W (f32) -> split-bf16 hi/lo planes in WMMA A-fragment order.
// One block per K-chunk; chunk blob is contiguous (hi plane, then lo plane).
// ---------------------------------------------------------------------------
__global__ __launch_bounds__(256)
void pack_w_kernel(const float* __restrict__ W, unsigned int* __restrict__ wp)
{
    const int kc  = blockIdx.x;
    const int tid = threadIdx.x;
    const int wl  = tid >> 3;            // fragment lane 0..31
    const int wv  = tid & 7;             // fragment VGPR 0..7
    const int wo  = wl & 15;             // M within tile
    const int wh  = wl >> 4;
    const int c   = kc * KC + a_frag_kloc(wh, wv);

    unsigned int* dst = wp + (size_t)kc * WCHUNK_WORDS;
#pragma unroll
    for (int t2 = 0; t2 < MT; ++t2) {
        int    o  = t2 * 16 + wo;
        float2 w2 = *(const float2*)(W + (size_t)o * CIN + c);
        unsigned int hi, lo;
        split2(w2.x, w2.y, hi, lo);
        int sa = (t2 * 32 + wl) * 8 + wv;
        dst[sa]               = hi;
        dst[sa + WFRAG_WORDS] = lo;
    }
}

// ---------------------------------------------------------------------------
// Main GEMM kernel. USE_WS=true: W tiles come pre-packed from workspace (TDM
// DMA when the builtin exists). USE_WS=false: self-contained (converts W in
// the loop, as in the previous round).
// ---------------------------------------------------------------------------
template <bool USE_WS>
__global__ __launch_bounds__(256)
void bn_relu_conv1x1_wmma(const float* __restrict__ x,
                          const float* __restrict__ gamma,
                          const float* __restrict__ beta,
                          const float* __restrict__ rmean,
                          const float* __restrict__ rvar,
                          const float* __restrict__ W,
                          const unsigned int* __restrict__ wp,
                          float* __restrict__ out)
{
    // --- LDS ---------------------------------------------------------------
    __shared__ float s_sc[CIN];                                   // 8448 B
    __shared__ float s_sh[CIN];                                   // 8448 B
    // x stage, packed bf16 pairs in B-fragment order: [v(0..7)][nt*32 + lane]
    __shared__ __align__(16) unsigned int s_xhi[8 * 256];         // 8192 B
    __shared__ __align__(16) unsigned int s_xlo[8 * 256];         // 8192 B
    // W stage: hi plane [0, 3072), lo plane [3072, 6144), A-fragment order
    __shared__ __align__(16) unsigned int s_w[WCHUNK_WORDS];      // 24576 B
    // total 57856 B

    const int tid  = threadIdx.x;
    const int lane = tid & 31;                 // wave32
    const int wave = tid >> 5;                 // 0..7
    const int n0   = blockIdx.x * TN;

    // --- BN scale/shift once per block -------------------------------------
    for (int c = tid; c < CIN; c += 256) {
        float inv = rsqrtf(rvar[c] + EPS);
        float sc  = gamma[c] * inv;
        s_sc[c]   = sc;
        s_sh[c]   = beta[c] - rmean[c] * sc;
    }

    // --- per-thread x-stage constants (n fixed across chunks) --------------
    const int  xn = tid & 127;                 // column within tile
    const int  th = tid >> 7;                  // 0/1: even/odd k-pair rows
    const int  gn = n0 + xn;                   // global column (always valid)
    const int  xb = gn / HW;
    const int  xs = gn - xb * HW;
    const long xbase = (long)xb * CIN * HW + xs;
    int wa[8];
#pragma unroll
    for (int i = 0; i < 8; ++i) {
        int j    = 2 * i + th;                 // k-pair row 0..15 (k = 2j, 2j+1)
        int v    = (j & 3) + 4 * (j >> 3);
        int half = (j >> 2) & 1;
        wa[i] = v * 256 + (xn >> 4) * 32 + (xn & 15) + 16 * half;
    }

    // --- per-thread W-stage constants (self-contained path) ----------------
    const int wl  = tid >> 3;
    const int wv  = tid & 7;
    const int wo  = wl & 15;
    const int wh  = wl >> 4;
    const int wkl = a_frag_kloc(wh, wv);

    // --- wave tiling: 4 M-groups x 2 N-groups ------------------------------
    const int mt0 = (wave & 3) * 3;            // 3 M tiles per wave
    const int nt0 = (wave >> 2) * 4;           // 4 N tiles per wave

    v8f acc[3][4] = {};

    __syncthreads();

    for (int kc = 0; kc < NCH; ++kc) {
        const int cb = kc * KC;

        // ---- stage W tile ------------------------------------------------
        if constexpr (USE_WS) {
#if defined(HAVE_TDM)
            if (wave == 0) {
                // D# for a 1-D 24576-byte DMA: wp[kc] blob -> s_w.
                unsigned lds_off = (unsigned)(size_t)(void*)s_w;   // addr[31:0] = LDS offset
                unsigned long long ga =
                    (unsigned long long)(size_t)(wp + (size_t)kc * WCHUNK_WORDS);
                u32x4 g0 = { 1u,                                  // count=1, user mode
                             lds_off,                             // lds_addr
                             (unsigned)ga,                        // global_addr[31:0]
                             (unsigned)((ga >> 32) & 0x1FFFFFFu)  // global_addr[56:32]
                                 | (2u << 30) };                  // type=2 ("image")
                i32x8 g1 = { (int)(3u << 16),          // data_size=8B, wg_mask=0
                             (int)(3072u << 16),       // tensor_dim0[15:0]=3072
                             (int)(1u << 16),          // tensor_dim0[31:16]=0, tensor_dim1=1
                             (int)(3072u << 16),       // tile_dim0=3072 (x 8B = 24576 B)
                             0,                        // tile_dim1=0, tile_dim2=0
                             3072,                     // tensor_dim0_stride[31:0]
                             0, 0 };
                i32x4 z4 = { 0, 0, 0, 0 };
                i32x8 z8 = { 0, 0, 0, 0, 0, 0, 0, 0 };
                // clang-23 / amdgpu-toolchain form: 6 args (g0,g1,g2,g3,g4,cpol)
                __builtin_amdgcn_tensor_load_to_lds(g0, g1, z4, z4, z8, 0);
            }
#else
            // Vectorized plain copy fallback: 1536 x 16B chunks, 6 per thread.
#pragma unroll
            for (int i = 0; i < 6; ++i) {
                int idx = i * 256 + tid;
                v4u t = *(const v4u*)(wp + (size_t)kc * WCHUNK_WORDS + (size_t)idx * 4);
                *(v4u*)&s_w[idx * 4] = t;
            }
#endif
        } else {
            // Self-contained: convert W in-loop.
#pragma unroll
            for (int t2 = 0; t2 < MT; ++t2) {
                int    o  = t2 * 16 + wo;
                int    c  = cb + wkl;
                float2 w2 = *(const float2*)(W + (long)o * CIN + c);
                unsigned int hi, lo;
                split2(w2.x, w2.y, hi, lo);
                int sa = (t2 * 32 + wl) * 8 + wv;
                s_w[sa]               = hi;
                s_w[sa + WFRAG_WORDS] = lo;
            }
        }

        // ---- stage x (32 x 128 f32 -> BN+ReLU -> split bf16 pairs) --------
#pragma unroll
        for (int i = 0; i < 8; ++i) {
            int   c  = cb + 2 * (2 * i + th);
            float g0 = x[xbase + (long)c * HW];
            float g1 = x[xbase + (long)(c + 1) * HW];
            float h0 = fmaxf(fmaf(g0, s_sc[c],     s_sh[c]),     0.f);
            float h1 = fmaxf(fmaf(g1, s_sc[c + 1], s_sh[c + 1]), 0.f);
            unsigned int hi, lo;
            split2(h0, h1, hi, lo);
            s_xhi[wa[i]] = hi;
            s_xlo[wa[i]] = lo;
            if (kc + 1 < NCH)                  // pull next chunk toward L2/L0
                __builtin_prefetch(&x[xbase + (long)(c + KC) * HW], 0, 0);
        }

#if defined(HAVE_TDM)
        if constexpr (USE_WS) {
            if (wave == 0) __builtin_amdgcn_s_wait_tensorcnt(0);
        }
#endif
        __syncthreads();

        // ---- load A fragments (this wave's 3 M tiles) ---------------------
        Frag ahi[3], alo[3];
#pragma unroll
        for (int m = 0; m < 3; ++m) {
            int base = ((mt0 + m) * 32 + lane) * 8;
            ahi[m].q[0] = *(const v4u*)&s_w[base];
            ahi[m].q[1] = *(const v4u*)&s_w[base + 4];
            alo[m].q[0] = *(const v4u*)&s_w[WFRAG_WORDS + base];
            alo[m].q[1] = *(const v4u*)&s_w[WFRAG_WORDS + base + 4];
        }

        // ---- B fragments + WMMA -------------------------------------------
#pragma unroll
        for (int t = 0; t < 4; ++t) {
            Frag bhi, blo;
#pragma unroll
            for (int v = 0; v < 8; ++v) {
                int a = v * 256 + (nt0 + t) * 32 + lane;
                bhi.w[v] = s_xhi[a];
                blo.w[v] = s_xlo[a];
            }
#pragma unroll
            for (int m = 0; m < 3; ++m) {
                acc[m][t] = __builtin_amdgcn_wmma_f32_16x16x32_bf16(
                    false, ahi[m].v, false, bhi.v, (short)0, acc[m][t], false, false);
                acc[m][t] = __builtin_amdgcn_wmma_f32_16x16x32_bf16(
                    false, ahi[m].v, false, blo.v, (short)0, acc[m][t], false, false);
                acc[m][t] = __builtin_amdgcn_wmma_f32_16x16x32_bf16(
                    false, alo[m].v, false, bhi.v, (short)0, acc[m][t], false, false);
            }
        }

        __syncthreads();   // protect LDS before next chunk's stores
    }

    // ---- write out[b, o, s]  (C/D: VGPR r -> M=r (lanes 0-15) / M=r+8) -----
    const int lhalf = lane >> 4;
#pragma unroll
    for (int t = 0; t < 4; ++t) {
        int  n = n0 + (nt0 + t) * 16 + (lane & 15);
        int  b = n / HW;
        int  s = n - b * HW;
        long obase = (long)b * COUT * HW + s;
#pragma unroll
        for (int m = 0; m < 3; ++m) {
            int o0 = (mt0 + m) * 16 + 8 * lhalf;
#pragma unroll
            for (int r = 0; r < 8; ++r)
                out[obase + (long)(o0 + r) * HW] = acc[m][t][r];
        }
    }
}

extern "C" void kernel_launch(void* const* d_in, const int* in_sizes, int n_in,
                              void* d_out, int out_size, void* d_ws, size_t ws_size,
                              hipStream_t stream) {
    const float* x     = (const float*)d_in[0];
    const float* gamma = (const float*)d_in[1];
    const float* beta  = (const float*)d_in[2];
    const float* rmean = (const float*)d_in[3];
    const float* rvar  = (const float*)d_in[4];
    const float* W     = (const float*)d_in[5];
    float*       out   = (float*)d_out;
    (void)in_sizes; (void)n_in; (void)out_size;

    if (ws_size >= WS_NEEDED && d_ws != nullptr) {
        unsigned int* wp = (unsigned int*)d_ws;
        pack_w_kernel<<<NCH, 256, 0, stream>>>(W, wp);
        bn_relu_conv1x1_wmma<true><<<NWG, 256, 0, stream>>>(
            x, gamma, beta, rmean, rvar, W, wp, out);
    } else {
        bn_relu_conv1x1_wmma<false><<<NWG, 256, 0, stream>>>(
            x, gamma, beta, rmean, rvar, W, nullptr, out);
    }
}
